// SetTransformer_48567490183858
// MI455X (gfx1250) — matrix operand order repaired and checked by reading, not compiled, not run on hardware
//
#include <hip/hip_runtime.h>

// ---------------------------------------------------------------------------
// CDNA5 (gfx1250) Set Transformer forward.
// Matmuls via v_wmma_f32_16x16x32_bf16 (wave32 WMMA); flash attention with
// online softmax; double-buffered LDS staging using gfx1250 async
// global->LDS loads when the toolchain exposes them.
// ---------------------------------------------------------------------------

typedef __bf16 bf16_t;
typedef bf16_t v16bf __attribute__((ext_vector_type(16)));
typedef bf16_t v8bf  __attribute__((ext_vector_type(8)));
typedef float  v8f   __attribute__((ext_vector_type(8)));

union FragBf { v16bf v; v8bf h[2]; unsigned short s[16]; };
union FragF  { v8f v; float f[8]; };

// ---- gfx1250 async global->LDS (ASYNCcnt) with sync fallback --------------
#if defined(__has_builtin)
#if __has_builtin(__builtin_amdgcn_global_load_async_to_lds_b128)
#define USE_ASYNC_LDS 1
#endif
#endif

#if defined(USE_ASYNC_LDS)
typedef int v4i_vs __attribute__((vector_size(16)));
typedef __attribute__((address_space(1))) v4i_vs gv4i;
typedef __attribute__((address_space(3))) v4i_vs lv4i;
__device__ __forceinline__ void async_b128(const unsigned short* g,
                                           unsigned short* l) {
  __builtin_amdgcn_global_load_async_to_lds_b128(
      (gv4i*)(void*)g, (lv4i*)(void*)l, 0, 0);
}
#if __has_builtin(__builtin_amdgcn_s_wait_asynccnt)
#define WAIT_ASYNC(n) __builtin_amdgcn_s_wait_asynccnt(n)
#else
#define WAIT_ASYNC(n) asm volatile("s_wait_asynccnt %0" ::"i"(n) : "memory")
#endif
#endif

__device__ __forceinline__ v8f vzero() {
  FragF z;
#pragma unroll
  for (int i = 0; i < 8; i++) z.f[i] = 0.0f;
  return z.v;
}

// f32 -> bf16 round-to-nearest-even
__device__ __forceinline__ unsigned short f2bf(float f) {
  unsigned int u = __float_as_uint(f);
  u += 0x7fffu + ((u >> 16) & 1u);
  return (unsigned short)(u >> 16);
}
__device__ __forceinline__ float bf2f(unsigned short s) {
  return __uint_as_float(((unsigned int)s) << 16);
}

// Build a 16xbf16 A/B fragment from two 16-byte chunks.
__device__ __forceinline__ v16bf make_frag(const unsigned short* p0,
                                           const unsigned short* p1) {
  FragBf r;
  r.h[0] = *(const v8bf*)p0;
  r.h[1] = *(const v8bf*)p1;
  return r.v;
}

#define LDS_STR 40  // 32 bf16 + 8 pad -> 80B row stride, conflict-free b128

// ---------------------------------------------------------------------------
// GEMM: out = act(A[M,512] @ W[N,512]^T + bias) (+ resid), dual f32/bf16 out.
// Block tile 128(M) x 128(N), 8 waves (4x2), wave tile 32x64 = 2x4 WMMA.
// Double-buffered LDS K-slabs; async global->LDS staging when available.
// ---------------------------------------------------------------------------
__global__ __launch_bounds__(256) void k_gemm(
    const unsigned short* __restrict__ A,   // [M,512] bf16
    const unsigned short* __restrict__ W,   // [N,512] bf16 (row-major weight)
    const float* __restrict__ bias,         // [N] or null
    const float* __restrict__ resid,        // [M,N] f32 or null
    float* __restrict__ outF,               // [M,N] f32 or null
    unsigned short* __restrict__ outB,      // [M,N] bf16 or null
    int M, int N, int relu)
{
  __shared__ unsigned short ldsA[2][128 * LDS_STR];
  __shared__ unsigned short ldsB[2][128 * LDS_STR];
  const int K = 512;
  const int m0 = blockIdx.y * 128;
  const int n0 = blockIdx.x * 128;
  const int t = threadIdx.x;
  const int lane = t & 31;
  const int wave = t >> 5;
  const int wm = (wave & 3) * 32;   // 4 waves along M
  const int wn = (wave >> 2) * 64;  // 2 waves along N
  const int l15 = lane & 15;
  const int g = lane >> 4;

  v8f acc[2][4];
#pragma unroll
  for (int i = 0; i < 2; i++)
#pragma unroll
    for (int j = 0; j < 4; j++) acc[i][j] = vzero();

  // staging map: thread -> (row, 16-col half); A and B tiles are 128x32
  const int sr = t >> 1, sc = (t & 1) * 16;
  const unsigned short* gA = A + (size_t)(m0 + sr) * K + sc;
  const unsigned short* gW = W + (size_t)(n0 + sr) * K + sc;
  unsigned short* sAp[2] = { ldsA[0] + sr * LDS_STR + sc,
                             ldsA[1] + sr * LDS_STR + sc };
  unsigned short* sBp[2] = { ldsB[0] + sr * LDS_STR + sc,
                             ldsB[1] + sr * LDS_STR + sc };

  auto stage = [&](int k0, int buf) {
#if defined(USE_ASYNC_LDS)
    async_b128(gA + k0,     sAp[buf]);
    async_b128(gA + k0 + 8, sAp[buf] + 8);
    async_b128(gW + k0,     sBp[buf]);
    async_b128(gW + k0 + 8, sBp[buf] + 8);
#else
    const uint4* pa = (const uint4*)(gA + k0);
    const uint4* pw = (const uint4*)(gW + k0);
    uint4 a0 = pa[0], a1 = pa[1];
    uint4 w0 = pw[0], w1 = pw[1];
    ((uint4*)sAp[buf])[0] = a0;
    ((uint4*)sAp[buf])[1] = a1;
    ((uint4*)sBp[buf])[0] = w0;
    ((uint4*)sBp[buf])[1] = w1;
#endif
  };

  const int nsteps = K / 32;
  stage(0, 0);
  for (int s = 0; s < nsteps; s++) {
    const int buf = s & 1;
    const bool more = (s + 1) < nsteps;
    if (more) {
      stage((s + 1) * 32, buf ^ 1);
      if (s + 2 < nsteps)
        __builtin_prefetch((const void*)(gA + (s + 2) * 32), 0, 1);
    }
#if defined(USE_ASYNC_LDS)
    if (more) { WAIT_ASYNC(4); } else { WAIT_ASYNC(0); }
#endif
    __syncthreads();

    const unsigned short* lA = ldsA[buf];
    const unsigned short* lB = ldsB[buf];
    v16bf af[2], bf[4];
#pragma unroll
    for (int i = 0; i < 2; i++) {
      // A layout: M = lane&15, K = 16g..16g+15 contiguous
      const unsigned short* pa = lA + (wm + i * 16 + l15) * LDS_STR;
      af[i] = make_frag(pa + 16 * g, pa + 16 * g + 8);
    }
#pragma unroll
    for (int j = 0; j < 4; j++) {
      // B layout: N = lane&15, K = 16g..16g+15 contiguous
      const unsigned short* pb = lB + (wn + j * 16 + l15) * LDS_STR + 16 * g;
      bf[j] = make_frag(pb, pb + 8);
    }
#pragma unroll
    for (int mi = 0; mi < 2; mi++)
#pragma unroll
      for (int ni = 0; ni < 4; ni++)
        acc[mi][ni] = __builtin_amdgcn_wmma_f32_16x16x32_bf16(
            false, af[mi], false, bf[ni], (short)0, acc[mi][ni], false, false);
    __syncthreads();
  }

  // epilogue: C layout vgpr r, lanes0-15 M=r, lanes16-31 M=r+8, N=lane&15
  const int hi = g * 8;
#pragma unroll
  for (int mi = 0; mi < 2; mi++)
#pragma unroll
    for (int ni = 0; ni < 4; ni++) {
      FragF c; c.v = acc[mi][ni];
      int col = n0 + wn + ni * 16 + l15;
      float bv = bias ? bias[col] : 0.0f;
#pragma unroll
      for (int r = 0; r < 8; r++) {
        int row = m0 + wm + mi * 16 + hi + r;
        float v = c.f[r] + bv;
        if (relu) v = v > 0.0f ? v : 0.0f;
        if (resid) v += resid[(size_t)row * N + col];
        if (outF) outF[(size_t)row * N + col] = v;
        if (outB) outB[(size_t)row * N + col] = f2bf(v);
      }
    }
  (void)M;
}

// ---------------------------------------------------------------------------
// Flash attention for one (batch, head, 16-query tile) per wave.
// O = Qp + softmax(Qp Kp^T * scale) @ Vp, heads of 64 within 512 cols.
// ---------------------------------------------------------------------------
__global__ __launch_bounds__(32) void k_attn(
    const unsigned short* __restrict__ Q,   // [B*Sq,512] bf16
    const unsigned short* __restrict__ Kp,  // [B*Sk,512] bf16
    const unsigned short* __restrict__ Vp,  // [B*Sk,512] bf16
    float* __restrict__ O,                  // [B*Sq,512] f32
    int B, int Sq, int Sk, float scale)
{
  __shared__ unsigned short ldsP[16 * LDS_STR];  // P tile 16x32 (padded)
  __shared__ unsigned short ldsV[32 * 72];       // V tile 32x64 (padded)
  const int lane = threadIdx.x;
  const int l15 = lane & 15;
  const int g = lane >> 4;
  const int qtiles = Sq >> 4;
  const int bid = blockIdx.x;
  const int qt = bid % qtiles;
  const int h = (bid / qtiles) & 7;
  const int b = bid / (qtiles * 8);
  const int hc = h * 64;
  const int q0 = qt * 16;

  // Q A-fragments (16x32 each, two k-steps over dh=64), kept in registers
  v16bf qa[2];
  {
    const unsigned short* qr = Q + (size_t)(b * Sq + q0 + l15) * 512 + hc;
#pragma unroll
    for (int kk = 0; kk < 2; kk++)
      qa[kk] = make_frag(qr + kk * 32 + 16 * g, qr + kk * 32 + 16 * g + 8);
  }

  v8f o[4];
#pragma unroll
  for (int i = 0; i < 4; i++) o[i] = vzero();
  float mrow[8], lrow[8];
#pragma unroll
  for (int r = 0; r < 8; r++) { mrow[r] = -1e30f; lrow[r] = 0.0f; }

  for (int kb = 0; kb < Sk; kb += 32) {
    { // stage V tile [32 keys][64 dh] -> LDS (clamped rows)
      int key = kb + lane; if (key > Sk - 1) key = Sk - 1;
      const unsigned short* vg = Vp + (size_t)(b * Sk + key) * 512 + hc;
      unsigned short* vs = ldsV + lane * 72;
#if defined(USE_ASYNC_LDS)
#pragma unroll
      for (int c = 0; c < 8; c++) async_b128(vg + c * 8, vs + c * 8);
#else
#pragma unroll
      for (int c = 0; c < 8; c++)
        ((uint4*)vs)[c] = ((const uint4*)vg)[c];
#endif
    }

    // scores S = Q K^T for 32 keys (two 16-key WMMA columns)
    FragF sfr[2];
#pragma unroll
    for (int nt = 0; nt < 2; nt++) {
      int key = kb + nt * 16 + l15;
      int keyc = key < Sk ? key : Sk - 1;
      v8f accs = vzero();
      const unsigned short* kr = Kp + (size_t)(b * Sk + keyc) * 512 + hc;
#pragma unroll
      for (int kk = 0; kk < 2; kk++) {
        const unsigned short* p = kr + kk * 32 + 16 * g;  // B frag: K contig
        v16bf bfrag = make_frag(p, p + 8);
        accs = __builtin_amdgcn_wmma_f32_16x16x32_bf16(
            false, qa[kk], false, bfrag, (short)0, accs, false, false);
      }
      sfr[nt].v = accs * scale;
      if (key >= Sk) {
#pragma unroll
        for (int r = 0; r < 8; r++) sfr[nt].f[r] = -1e30f;  // masked column
      }
    }

    // online softmax: row max / rescale / exp / row sum (16-lane groups)
    float mnew[8], alpha[8];
#pragma unroll
    for (int r = 0; r < 8; r++) {
      float v = fmaxf(sfr[0].f[r], sfr[1].f[r]);
#pragma unroll
      for (int off = 1; off < 16; off <<= 1) v = fmaxf(v, __shfl_xor(v, off, 32));
      mnew[r] = fmaxf(mrow[r], v);
      alpha[r] = __expf(mrow[r] - mnew[r]);
      mrow[r] = mnew[r];
    }
#pragma unroll
    for (int nt = 0; nt < 2; nt++)
#pragma unroll
      for (int r = 0; r < 8; r++)
        sfr[nt].f[r] = __expf(sfr[nt].f[r] - mnew[r]);
#pragma unroll
    for (int r = 0; r < 8; r++) {
      float sum = sfr[0].f[r] + sfr[1].f[r];
#pragma unroll
      for (int off = 1; off < 16; off <<= 1) sum += __shfl_xor(sum, off, 32);
      lrow[r] = lrow[r] * alpha[r] + sum;
    }
#pragma unroll
    for (int i = 0; i < 4; i++) {
      FragF of; of.v = o[i];
#pragma unroll
      for (int r = 0; r < 8; r++) of.f[r] *= alpha[r];
      o[i] = of.v;
    }

    // relayout P (C layout) -> A fragment via LDS
#pragma unroll
    for (int nt = 0; nt < 2; nt++)
#pragma unroll
      for (int r = 0; r < 8; r++)
        ldsP[(r + g * 8) * LDS_STR + nt * 16 + l15] = f2bf(sfr[nt].f[r]);
    asm volatile("s_wait_dscnt 0" ::: "memory");
#if defined(USE_ASYNC_LDS)
    WAIT_ASYNC(0);  // V tile resident before fragment reads
#endif

    const unsigned short* pp = ldsP + l15 * LDS_STR;
    v16bf pfrag = make_frag(pp + 16 * g, pp + 16 * g + 8);

    // O += P @ V over 4 dh-column tiles
#pragma unroll
    for (int nt4 = 0; nt4 < 4; nt4++) {
      FragBf vf;
#pragma unroll
      for (int i = 0; i < 16; i++)
        vf.s[i] = ldsV[(16 * g + i) * 72 + nt4 * 16 + l15];
      o[nt4] = __builtin_amdgcn_wmma_f32_16x16x32_bf16(
          false, pfrag, false, vf.v, (short)0, o[nt4], false, false);
    }
  }

  // epilogue: normalize by l, add Q residual, store f32
#pragma unroll
  for (int nt4 = 0; nt4 < 4; nt4++) {
    FragF of; of.v = o[nt4];
#pragma unroll
    for (int r = 0; r < 8; r++) {
      int row = b * Sq + q0 + g * 8 + r;
      int col = hc + nt4 * 16 + l15;
      float v = of.f[r] / lrow[r] + bf2f(Q[(size_t)row * 512 + col]);
      O[(size_t)row * 512 + col] = v;
    }
  }
  (void)B;
}

// ---------------------------------------------------------------------------
// LayerNorm over 512 cols; dual f32/bf16 outputs.
// ---------------------------------------------------------------------------
__global__ __launch_bounds__(128) void k_ln(
    const float* __restrict__ Xin, const float* __restrict__ gamma,
    const float* __restrict__ beta, float* __restrict__ outF,
    unsigned short* __restrict__ outB)
{
  __shared__ float red[8];
  const int row = blockIdx.x, t = threadIdx.x;
  const float* xr = Xin + (size_t)row * 512;
  float v[4]; float s = 0.0f;
#pragma unroll
  for (int i = 0; i < 4; i++) { v[i] = xr[t + i * 128]; s += v[i]; }
#pragma unroll
  for (int o = 1; o < 32; o <<= 1) s += __shfl_xor(s, o, 32);
  if ((t & 31) == 0) red[t >> 5] = s;
  __syncthreads();
  float mu = (red[0] + red[1] + red[2] + red[3]) * (1.0f / 512.0f);
  float q = 0.0f;
#pragma unroll
  for (int i = 0; i < 4; i++) { float d = v[i] - mu; q += d * d; }
#pragma unroll
  for (int o = 1; o < 32; o <<= 1) q += __shfl_xor(q, o, 32);
  if ((t & 31) == 0) red[4 + (t >> 5)] = q;
  __syncthreads();
  float var = (red[4] + red[5] + red[6] + red[7]) * (1.0f / 512.0f);
  float rstd = rsqrtf(var + 1e-5f);
#pragma unroll
  for (int i = 0; i < 4; i++) {
    int c = t + i * 128;
    float y = (v[i] - mu) * rstd * gamma[c] + beta[c];
    if (outF) outF[(size_t)row * 512 + c] = y;
    if (outB) outB[(size_t)row * 512 + c] = f2bf(y);
  }
}

// ---------------------------------------------------------------------------
// f32 -> bf16 convert; batched broadcast-convert (inducing points / seeds).
// ---------------------------------------------------------------------------
__global__ void k_cvt(const float* __restrict__ src,
                      unsigned short* __restrict__ dst, int n) {
  int i = blockIdx.x * blockDim.x + threadIdx.x;
  if (i < n) dst[i] = f2bf(src[i]);
}
__global__ void k_bcast(const float* __restrict__ src,
                        unsigned short* __restrict__ dst, int per, int total) {
  int i = blockIdx.x * blockDim.x + threadIdx.x;
  if (i < total) dst[i] = f2bf(src[i % per]);
}

// ---------------------------------------------------------------------------
// Host orchestration.
// ---------------------------------------------------------------------------
extern "C" void kernel_launch(void* const* d_in, const int* in_sizes, int n_in,
                              void* d_out, int out_size, void* d_ws,
                              size_t ws_size, hipStream_t stream) {
  (void)in_sizes; (void)n_in; (void)out_size; (void)ws_size;
  int p = 0;
  auto f = [&](int i) { return (const float*)d_in[i]; };
  struct MabP { const float *Wq,*bq,*Wk,*bk,*Wv,*bv,*Wo,*bo,*g0,*b0,*g1,*b1; };
  auto getMab = [&]() {
    MabP m;
    m.Wq = f(p++); m.bq = f(p++); m.Wk = f(p++); m.bk = f(p++);
    m.Wv = f(p++); m.bv = f(p++); m.Wo = f(p++); m.bo = f(p++);
    m.g0 = f(p++); m.b0 = f(p++); m.g1 = f(p++); m.b1 = f(p++);
    return m;
  };
  const float* X = f(p++);
  const float* I0 = f(p++); MabP e0m0 = getMab(); MabP e0m1 = getMab();
  const float* I1 = f(p++); MabP e1m0 = getMab(); MabP e1m1 = getMab();
  const float* Sd = f(p++); MabP pma = getMab(); MabP sab1 = getMab();
  MabP sab2 = getMab();
  const float* Wout = f(p++); const float* bout = f(p++);

  const int Bb = 32, D = 512;
  const int MX = Bb * 2048;  // 65536

  char* base = (char*)d_ws; size_t off = 0;
  auto alloc = [&](size_t bytes) {
    void* r = base + off;
    off = (off + bytes + 255) & ~(size_t)255;
    return r;
  };
  struct MabW { unsigned short *Wq, *Wk, *Wv, *Wo; };
  const MabP* mabs[7] = { &e0m0, &e0m1, &e1m0, &e1m1, &pma, &sab1, &sab2 };
  MabW wb[7];
  for (int i = 0; i < 7; i++) {
    wb[i].Wq = (unsigned short*)alloc((size_t)D * D * 2);
    wb[i].Wk = (unsigned short*)alloc((size_t)D * D * 2);
    wb[i].Wv = (unsigned short*)alloc((size_t)D * D * 2);
    wb[i].Wo = (unsigned short*)alloc((size_t)D * D * 2);
  }
  unsigned short* WoutB = (unsigned short*)alloc((size_t)256 * D * 2);
  unsigned short* Xb  = (unsigned short*)alloc((size_t)MX * D * 2);
  unsigned short* Ib  = (unsigned short*)alloc((size_t)Bb * 64 * D * 2);
  unsigned short* Hb  = (unsigned short*)alloc((size_t)Bb * 64 * D * 2);
  unsigned short* Sb  = (unsigned short*)alloc((size_t)Bb * 16 * D * 2);
  unsigned short* Yb  = (unsigned short*)alloc((size_t)Bb * 16 * D * 2);
  unsigned short* Y2b = (unsigned short*)alloc((size_t)Bb * 16 * D * 2);
  unsigned short* Qp  = (unsigned short*)alloc((size_t)MX * D * 2);
  unsigned short* Kp  = (unsigned short*)alloc((size_t)MX * D * 2);
  unsigned short* Vp  = (unsigned short*)alloc((size_t)MX * D * 2);
  float* Of  = (float*)alloc((size_t)MX * D * 4);
  float* LNf = (float*)alloc((size_t)MX * D * 4);
  unsigned short* LNb = (unsigned short*)alloc((size_t)MX * D * 2);

  auto cvt = [&](const float* s, unsigned short* d, size_t n) {
    k_cvt<<<dim3((unsigned)((n + 255) / 256)), dim3(256), 0, stream>>>(s, d, (int)n);
  };
  for (int i = 0; i < 7; i++) {
    cvt(mabs[i]->Wq, wb[i].Wq, (size_t)D * D);
    cvt(mabs[i]->Wk, wb[i].Wk, (size_t)D * D);
    cvt(mabs[i]->Wv, wb[i].Wv, (size_t)D * D);
    cvt(mabs[i]->Wo, wb[i].Wo, (size_t)D * D);
  }
  cvt(Wout, WoutB, (size_t)256 * D);
  cvt(X, Xb, (size_t)MX * D);

  auto gemm = [&](const unsigned short* A, const unsigned short* W,
                  const float* bias, const float* resid, float* oF,
                  unsigned short* oB, int M, int N, int relu) {
    k_gemm<<<dim3(N / 128, M / 128), dim3(256), 0, stream>>>(
        A, W, bias, resid, oF, oB, M, N, relu);
  };
  auto ln = [&](const float* in, const float* g, const float* b, int M,
                float* oF, unsigned short* oB) {
    k_ln<<<dim3(M), dim3(128), 0, stream>>>(in, g, b, oF, oB);
  };
  auto bcast = [&](const float* s, unsigned short* d, int rows) {
    int per = rows * D, total = Bb * per;
    k_bcast<<<dim3((total + 255) / 256), dim3(256), 0, stream>>>(s, d, per, total);
  };
  const float scale = 0.044194173824159216f;  // 1/sqrt(512), per reference

  auto run_mab = [&](const unsigned short* Qin, int Sq, const unsigned short* Kin,
                     int Sk, const MabW& w, const MabP& m, unsigned short* outB) {
    int Mq = Bb * Sq, Mk = Bb * Sk;
    gemm(Qin, w.Wq, m.bq, nullptr, nullptr, Qp, Mq, D, 0);
    gemm(Kin, w.Wk, m.bk, nullptr, nullptr, Kp, Mk, D, 0);
    gemm(Kin, w.Wv, m.bv, nullptr, nullptr, Vp, Mk, D, 0);
    k_attn<<<dim3(Bb * 8 * (Sq / 16)), dim3(32), 0, stream>>>(
        Qp, Kp, Vp, Of, Bb, Sq, Sk, scale);
    ln(Of, m.g0, m.b0, Mq, LNf, LNb);
    gemm(LNb, w.Wo, m.bo, LNf, Of, nullptr, Mq, D, 1);  // resid + relu
    ln(Of, m.g1, m.b1, Mq, nullptr, outB);
  };

  // encoder: 2x ISAB
  bcast(I0, Ib, 64);
  run_mab(Ib, 64, Xb, 2048, wb[0], e0m0, Hb);
  run_mab(Xb, 2048, Hb, 64, wb[1], e0m1, Xb);
  bcast(I1, Ib, 64);
  run_mab(Ib, 64, Xb, 2048, wb[2], e1m0, Hb);
  run_mab(Xb, 2048, Hb, 64, wb[3], e1m1, Xb);
  // decoder: PMA + SAB + SAB
  bcast(Sd, Sb, 16);
  run_mab(Sb, 16, Xb, 2048, wb[4], pma, Yb);
  run_mab(Yb, 16, Yb, 16, wb[5], sab1, Y2b);
  run_mab(Y2b, 16, Y2b, 16, wb[6], sab2, Yb);
  // final linear -> d_out [32,16,256] f32
  gemm(Yb, WoutB, bout, nullptr, (float*)d_out, nullptr, Bb * 16, 256, 0);
}
